// Encoding_48919677501760
// MI455X (gfx1250) — matrix operand order repaired and checked by reading, not compiled, not run on hardware
//
#include <hip/hip_runtime.h>

// ---------------------------------------------------------------------------
// Encoding layer (NetVLAD-style), fused single-pass kernel for MI455X/gfx1250.
// B=8, D=128, K=32, N=H*W=4096.
//   cross[b,n,k] = sum_d X[b,d,n]*C[k,d]           (WMMA bf16 GEMM #1)
//   A = softmax_k( scale_k * (|x|^2 - 2*cross + |c_k|^2) )
//   E[b,k,d] = sum_n A[b,n,k]*X[b,d,n] - (sum_n A[b,n,k])*C[k,d]  (WMMA GEMM #2)
// Memory-bound: X read exactly once (16.8 MB), staged in LDS.
// All bf16 LDS arrays use a 272-byte row stride (16B aligned) so every WMMA
// fragment is exactly two ds_load_b128.
// ---------------------------------------------------------------------------

typedef __bf16 bf16_t;
typedef bf16_t v16bf __attribute__((ext_vector_type(16)));
typedef float  v8f   __attribute__((ext_vector_type(8)));
typedef unsigned int v4u __attribute__((ext_vector_type(4)));

#define D_DIM 128
#define K_DIM 32
#define N_TOT 4096
#define TN    128           // descriptors per block
#define NTHREADS 128        // 4 waves (wave32)

#define ROWB 272            // bf16 row stride in bytes (136 elems, 16B aligned)

// LDS layout (bytes)
#define OFF_XND   0          // bf16 Xtile[n][d]
#define OFF_XDN   34816      // bf16 Xt[d][n]
#define OFF_CW    69632      // bf16 C[k][d]
#define OFF_ATT   78336      // f32  att[n][k], row stride 36 floats
#define OFF_ATTT  96768      // bf16 attT[k][n]
#define OFF_XSQ   105472     // f32 [128]
#define OFF_CSQ   105984     // f32 [32]
#define OFF_SK    106112     // f32 [32]
#define SMEM_BYTES 106240

// float -> bf16 bits, round to nearest even
__device__ inline unsigned bfbits(float x) {
  unsigned u = __builtin_bit_cast(unsigned, x);
  return (u + 0x7FFFu + ((u >> 16) & 1u)) >> 16;
}
__device__ inline bf16_t f2bf(float x) {
  unsigned short s = (unsigned short)bfbits(x);
  return __builtin_bit_cast(bf16_t, s);
}

// A-matrix fragment (16x32 bf16, ISA layout): lane m: VGPR j<4 = K{2j,2j+1}+8*khi,
// VGPR j>=4 = K 16+{..}+8*khi  -> two aligned 16B runs at +0 and +32 bytes.
__device__ inline v16bf load_fragA(const char* rowbase, int byteoff) {
  union { v16bf h; v4u q[2]; } f;
  f.q[0] = *(const v4u*)(rowbase + byteoff);        // ds_load_b128
  f.q[1] = *(const v4u*)(rowbase + byteoff + 32);   // ds_load_b128
  return f.h;
}

// B-matrix fragment (32x16 bf16): same K order as A: K{2j,2j+1}+8*khi for j<4,
// 16+{..}+8*khi for j>=4  -> two aligned 16B runs at +0 and +32 bytes.
__device__ inline v16bf load_fragB(const char* rowbase, int byteoff) {
  union { v16bf h; v4u q[2]; } f;
  f.q[0] = *(const v4u*)(rowbase + byteoff);        // ds_load_b128
  f.q[1] = *(const v4u*)(rowbase + byteoff + 32);   // ds_load_b128
  return f.h;
}

__global__ void enc_zero_kernel(float* __restrict__ out, int n) {
  int i = blockIdx.x * blockDim.x + threadIdx.x;
  if (i < n) out[i] = 0.0f;
}

__global__ __launch_bounds__(NTHREADS)
void enc_fused_kernel(const float* __restrict__ Xg,
                      const float* __restrict__ Cg,
                      const float* __restrict__ Sg,
                      float* __restrict__ Out) {
  extern __shared__ char smem[];
  bf16_t* xdn  = (bf16_t*)(smem + OFF_XDN);
  float*  att  = (float*) (smem + OFF_ATT);
  bf16_t* attT = (bf16_t*)(smem + OFF_ATTT);
  float*  xsq  = (float*) (smem + OFF_XSQ);
  float*  csq  = (float*) (smem + OFF_CSQ);
  float*  sk   = (float*) (smem + OFF_SK);

  const int t    = threadIdx.x;
  const int b    = blockIdx.x >> 5;     // 32 n-tiles per batch
  const int tile = blockIdx.x & 31;
  const int n0   = tile * TN;

  // --- stage codewords (bf16): thread t -> row k=t/4, 32 contiguous d ---
  {
    const int k  = t >> 2;
    const int d0 = (t & 3) * 32;
    const float* cp = Cg + k * D_DIM + d0;
    char* dst = smem + OFF_CW + k * ROWB + d0 * 2;
#pragma unroll
    for (int c = 0; c < 4; ++c) {       // 4 x (8 bf16) = 32 elems
      v4u pk;
#pragma unroll
      for (int j = 0; j < 4; ++j) {
        float v0 = cp[c * 8 + 2 * j];
        float v1 = cp[c * 8 + 2 * j + 1];
        pk[j] = bfbits(v0) | (bfbits(v1) << 16);
      }
      *(v4u*)(dst + c * 16) = pk;       // ds_store_b128
    }
  }
  if (t < K_DIM) {
    float a = 0.f;
    for (int d = 0; d < D_DIM; ++d) { float v = Cg[t * D_DIM + d]; a += v * v; }
    csq[t] = a;
  }

  // --- stream X tile from HBM once (coalesced over n), dual LDS layout ---
  {
    const float* xp = Xg + (size_t)b * D_DIM * N_TOT + n0 + t;
    char* xrow = smem + OFF_XND + t * ROWB;   // thread t owns row n=t
    float a = 0.f;
    for (int dc = 0; dc < 16; ++dc) {         // 16 chunks of 8 d-values
      v4u pk;
#pragma unroll
      for (int j = 0; j < 4; ++j) {
        int d = dc * 8 + 2 * j;
        float v0 = xp[(size_t)d * N_TOT];
        float v1 = xp[(size_t)(d + 1) * N_TOT];
        a += v0 * v0 + v1 * v1;
        pk[j] = bfbits(v0) | (bfbits(v1) << 16);
        xdn[d * 136 + t]       = f2bf(v0);    // transposed copy (scalar b16)
        xdn[(d + 1) * 136 + t] = f2bf(v1);
      }
      *(v4u*)(xrow + dc * 16) = pk;           // ds_store_b128
    }
    xsq[t] = a;
  }
  __syncthreads();

  const int wave = t >> 5;
  const int lane = t & 31;
  const int m16  = lane & 15;
  const int khi  = lane >> 4;

  // --- pass 1: cross[n][k], 8 n-blocks x 2 k-blocks of 16x16, Kdim=128 ---
#pragma unroll
  for (int tt = wave * 4; tt < wave * 4 + 4; ++tt) {
    const int nb = (tt >> 1) * 16;
    const int kb = (tt & 1) * 16;
    const char* arow = smem + OFF_XND + (nb + m16) * ROWB;
    const char* brow = smem + OFF_CW  + (kb + m16) * ROWB;
    v8f acc = {};
#pragma unroll
    for (int db = 0; db < 4; ++db) {
      v16bf A  = load_fragA(arow, db * 64 + khi * 16);
      v16bf Bf = load_fragB(brow, db * 64 + khi * 16);
      acc = __builtin_amdgcn_wmma_f32_16x16x32_bf16(false, A, false, Bf,
                                                    (short)0, acc, false, false);
    }
    // C/D layout: VGPR r, lanes<16 -> M=r, lanes>=16 -> M=r+8; N=lane%16
#pragma unroll
    for (int r = 0; r < 8; ++r)
      att[(nb + r + 8 * khi) * 36 + kb + m16] = acc[r];
  }
  __syncthreads();

  // --- softmax over k (scale_k * dist2, faithful to reference: no negation) ---
  {
    const int n = t;
    const float x2 = xsq[n];
    float vk[K_DIM];
    float mx = -3.4e38f;
#pragma unroll
    for (int k = 0; k < K_DIM; ++k) {
      float cr = att[n * 36 + k];
      float z = (x2 - 2.f * cr + csq[k]) * Sg[k];
      vk[k] = z;
      mx = fmaxf(mx, z);
    }
    float sum = 0.f;
#pragma unroll
    for (int k = 0; k < K_DIM; ++k) { float e = __expf(vk[k] - mx); vk[k] = e; sum += e; }
    const float inv = 1.f / sum;
#pragma unroll
    for (int k = 0; k < K_DIM; ++k) {
      float p = vk[k] * inv;
      att[n * 36 + k] = p;              // f32 copy for deterministic s_k sum
      attT[k * 136 + n] = f2bf(p);      // bf16 transposed for pass-2 A
    }
  }
  __syncthreads();

  // --- s_k = sum_n A[n][k] (per-tile partial, deterministic) ---
  if (t < K_DIM) {
    float a = 0.f;
    for (int n = 0; n < TN; ++n) a += att[n * 36 + t];
    sk[t] = a;
  }
  __syncthreads();

  // --- pass 2: E_partial[k][d] = A^T * Xtile, 2 k-blocks x 8 d-blocks ---
#pragma unroll
  for (int tt = wave * 4; tt < wave * 4 + 4; ++tt) {
    const int kb = (tt >> 3) * 16;
    const int db = (tt & 7) * 16;
    const char* arow = smem + OFF_ATTT + (kb + m16) * ROWB;
    const char* brow = smem + OFF_XDN  + (db + m16) * ROWB;
    v8f acc = {};
#pragma unroll
    for (int nc = 0; nc < 4; ++nc) {
      v16bf A  = load_fragA(arow, nc * 64 + khi * 16);
      v16bf Bf = load_fragB(brow, nc * 64 + khi * 16);
      acc = __builtin_amdgcn_wmma_f32_16x16x32_bf16(false, A, false, Bf,
                                                    (short)0, acc, false, false);
    }
    const int d = db + m16;
#pragma unroll
    for (int r = 0; r < 8; ++r) {
      const int k = kb + r + 8 * khi;
      // fold this tile's -s_k * c[k,d] contribution; partials over 32 tiles sum
      float val = acc[r] - sk[k] * Cg[k * D_DIM + d];
      atomicAdd(&Out[(size_t)b * K_DIM * D_DIM + k * D_DIM + d], val);
    }
  }
}

extern "C" void kernel_launch(void* const* d_in, const int* in_sizes, int n_in,
                              void* d_out, int out_size, void* d_ws, size_t ws_size,
                              hipStream_t stream) {
  (void)in_sizes; (void)n_in; (void)d_ws; (void)ws_size;
  const float* X  = (const float*)d_in[0];   // [8,128,64,64]
  const float* Cw = (const float*)d_in[1];   // [32,128]
  const float* Sc = (const float*)d_in[2];   // [32]
  float* Out = (float*)d_out;                // [8,32,128]

  static bool attr_set = false;
  if (!attr_set) {
    (void)hipFuncSetAttribute((const void*)enc_fused_kernel,
                              hipFuncAttributeMaxDynamicSharedMemorySize,
                              SMEM_BYTES);
    attr_set = true;
  }

  const int total = out_size;                // 8*32*128 = 32768
  enc_zero_kernel<<<(total + 255) / 256, 256, 0, stream>>>(Out, total);
  enc_fused_kernel<<<8 * 32, NTHREADS, SMEM_BYTES, stream>>>(X, Cw, Sc, Out);
}
